// Hermite2d_50560355008807
// MI455X (gfx1250) — matrix operand order repaired and checked by reading, not compiled, not run on hardware
//
#include <hip/hip_runtime.h>
#include <hip/hip_bf16.h>

typedef __attribute__((ext_vector_type(16))) _Float16 v16h;
typedef __attribute__((ext_vector_type(8)))  _Float16 v8h;
typedef __attribute__((ext_vector_type(8)))  float    v8f;
typedef __attribute__((ext_vector_type(4)))  unsigned int u32x4;
typedef __attribute__((ext_vector_type(8)))  int          i32x8;
typedef __attribute__((ext_vector_type(4)))  int          i32x4;

#define HW    256
#define BATCH 2
#define CN    64
#define C2    128

#if __has_builtin(__builtin_amdgcn_tensor_load_to_lds) && \
    __has_builtin(__builtin_amdgcn_tensor_store_from_lds)
#define USE_TDM 1
#else
#define USE_TDM 0
#endif

// ---------------------------------------------------------------------------
// Tensor Data Mover descriptor builder (CDNA5 D# per CDNA5 ISA).
// 2D tile: dim0 = contiguous f16 elements (data_size=2B), dim1 = lines.
// Reads beyond tensor dims are hardware zero-filled; OOB writes are dropped.
// ---------------------------------------------------------------------------
#if USE_TDM
__device__ __forceinline__ void tdm_build(u32x4& g0, i32x8& g1,
                                          unsigned lds_addr, const void* gptr,
                                          unsigned tensor_d0, unsigned tensor_d1,
                                          unsigned tile_d0, unsigned tile_d1,
                                          unsigned stride_d0,
                                          unsigned long long stride_d1) {
  unsigned long long ga = (unsigned long long)(size_t)gptr;
  g0[0] = 1u;                                           // count=1 (valid), user mode
  g0[1] = lds_addr;                                     // LDS byte address
  g0[2] = (unsigned)(ga & 0xffffffffu);                 // global_addr[31:0]
  g0[3] = (unsigned)((ga >> 32) & 0x1ffffffu) | (2u << 30); // addr[56:32] | type=2
  g1[0] = (int)(1u << 16);                              // data_size=1 (2 bytes)
  g1[1] = (int)((tensor_d0 & 0xffffu) << 16);           // tensor_dim0[15:0]
  g1[2] = (int)((tensor_d0 >> 16) | ((tensor_d1 & 0xffffu) << 16));
  g1[3] = (int)((tensor_d1 >> 16) | (tile_d0 << 16));   // tile_dim0
  g1[4] = (int)(tile_d1 & 0xffffu);                     // tile_dim1 (tile_dim2=0)
  g1[5] = (int)stride_d0;                               // tensor_dim0_stride[31:0]
  g1[6] = (int)((unsigned)(stride_d1 & 0xffffu) << 16); // dim1_stride[15:0] hi half
  g1[7] = (int)((stride_d1 >> 16) & 0xffffffffu);       // dim1_stride[47:16]
}

__device__ __forceinline__ void tdm_load_2d(unsigned lds_addr, const void* gptr,
                                            unsigned tensor_d0, unsigned tensor_d1,
                                            unsigned tile_d0, unsigned tile_d1,
                                            unsigned stride_d0,
                                            unsigned long long stride_d1) {
  u32x4 g0; i32x8 g1;
  tdm_build(g0, g1, lds_addr, gptr, tensor_d0, tensor_d1, tile_d0, tile_d1,
            stride_d0, stride_d1);
  i32x4 gz = {0, 0, 0, 0};
#if defined(__clang_major__) && __clang_major__ >= 23
  i32x8 gz8 = {0, 0, 0, 0, 0, 0, 0, 0};
  __builtin_amdgcn_tensor_load_to_lds(g0, g1, gz, gz, gz8, 0);
#else
  __builtin_amdgcn_tensor_load_to_lds(g0, g1, gz, gz, 0);
#endif
}

// Contiguous 1D tile store: LDS -> global, n f16 elements (n < 65536).
__device__ __forceinline__ void tdm_store_1d(unsigned lds_addr, void* gptr,
                                             unsigned n) {
  u32x4 g0; i32x8 g1;
  tdm_build(g0, g1, lds_addr, gptr, n, 1u, n, 1u, n, (unsigned long long)n);
  i32x4 gz = {0, 0, 0, 0};
#if defined(__clang_major__) && __clang_major__ >= 23
  i32x8 gz8 = {0, 0, 0, 0, 0, 0, 0, 0};
  __builtin_amdgcn_tensor_store_from_lds(g0, g1, gz, gz, gz8, 0);
#else
  __builtin_amdgcn_tensor_store_from_lds(g0, g1, gz, gz, 0);
#endif
}
#endif  // USE_TDM

__device__ __forceinline__ void tdm_wait() {
#if __has_builtin(__builtin_amdgcn_s_wait_tensorcnt)
  __builtin_amdgcn_s_wait_tensorcnt(0);
#else
  asm volatile("s_wait_tensorcnt 0x0" ::: "memory");
#endif
}

// ---------------------------------------------------------------------------
// Kernel 0a: convert x (fp32 NCHW) -> f16 NHWC ([b][h][w][c], c contiguous)
// so conv staging is channel-contiguous (TDM dim0 / 128-bit loads).
// ---------------------------------------------------------------------------
__global__ __launch_bounds__(256) void prep_x_kernel(const float* __restrict__ x,
                                                     _Float16* __restrict__ xh) {
  int i = blockIdx.x * 256 + threadIdx.x;
  if (i >= BATCH * HW * HW * CN) return;
  int c = i & 63, w = (i >> 6) & 255, h = (i >> 14) & 255, b = i >> 22;
  xh[i] = (_Float16)x[(((size_t)(b * CN + c) * HW) + h) * HW + w];
}

// ---------------------------------------------------------------------------
// Kernel 0b: pre-swizzle conv weights (fp32 OIHW, K = (r*3+s)*Cin + c) into
// exact WMMA B-fragment order: frag[kstep][ntile(8)][lane(32)][e(16)],
// element e of lane L holds B(K = kstep*32 + (L>>4)*8 + e + (e>=8 ? 8 : 0), N = ntile*16 + (L&15)).
// ---------------------------------------------------------------------------
__global__ __launch_bounds__(256) void prep_w_kernel(const float* __restrict__ w,
                                                     _Float16* __restrict__ frag,
                                                     int cin, int taps) {
  int t = blockIdx.x * 256 + threadIdx.x;
  int ksteps = (cin * taps) / 32;
  if (t >= ksteps * 8 * 32) return;
  int lane = t & 31, nt = (t >> 5) & 7, ks = t >> 8;
  int kh = lane >> 4, n = nt * 16 + (lane & 15);
  #pragma unroll
  for (int e = 0; e < 16; ++e) {
    int K  = ks * 32 + kh * 8 + e + (e >= 8 ? 8 : 0);
    int rs = K / cin, c = K - rs * cin;
    float val = (taps == 9) ? w[((size_t)(n * cin + c)) * 9 + rs]
                            : w[(size_t)n * cin + c];
    frag[(size_t)t * 16 + e] = (_Float16)val;
  }
}

// ---------------------------------------------------------------------------
// 3x3 conv (SAME) as implicit GEMM with v_wmma_f32_16x16x32_f16.
// Tile: one (b,h) row segment of 64 pixels x all 128 out-channels.
// 8 waves = 4 M-waves x 2 N-waves; 4 accumulators (16x16) per wave.
// In:  halo slab (3 x 66 x CIN f16) DMA'd to LDS by the TDM (one 2D descriptor
//      per row; right-edge halo zero-filled in hardware via tensor_dim1 clamp).
// Out: BN(+ReLU) tile staged in LDS, one contiguous 16 KB tensor_store_from_lds.
// ---------------------------------------------------------------------------
template <int CIN>
__global__ __launch_bounds__(256) void conv3x3_bn_kernel(
    const _Float16* __restrict__ xin,    // [B][H][W][CIN] f16
    const _Float16* __restrict__ wfrag,  // swizzled B fragments
    const float* __restrict__ gg, const float* __restrict__ bbeta,
    const float* __restrict__ mmean, const float* __restrict__ vvar,
    _Float16* __restrict__ outp,         // [B][H][W][128] f16
    int do_relu) {
  __shared__ __align__(16) _Float16 sA[3 * 66 * CIN];
#if USE_TDM
  __shared__ __align__(16) _Float16 sOut[64 * C2];
#endif

  const int tid = threadIdx.x;
  const int blk = blockIdx.x;
  const int b = blk >> 10, rem = blk & 1023;
  const int h = rem >> 2, w0 = (rem & 3) << 6;
  const int CV = CIN / 8;          // v8h vectors per (row, w)

#if USE_TDM
  // Manual zero-fill only where the TDM cannot (negative-coordinate halo).
  {
    v8h z = {};
    #pragma unroll
    for (int r = 0; r < 3; ++r) {
      int hh = h - 1 + r;
      if ((unsigned)hh >= HW)                      // OOB input row
        for (int v = tid; v < 66 * CV; v += 256)
          *(v8h*)(&sA[r * 66 * CIN + v * 8]) = z;
    }
    if (w0 == 0)                                    // left-edge halo column
      for (int v = tid; v < 3 * CV; v += 256) {
        int r = v / CV, cv = v - r * CV;
        *(v8h*)(&sA[(r * 66) * CIN + cv * 8]) = z;
      }
  }
  if (tid < 32) {                                   // wave 0 drives the TDM
    const bool ledge = (w0 == 0);
    const int wstart = w0 - 1 + (ledge ? 1 : 0);
    const unsigned wcnt = ledge ? 65u : 66u;
    const unsigned ldsbase = (unsigned)(size_t)&sA[0];
    for (int r = 0; r < 3; ++r) {
      int hh = h - 1 + r;
      if ((unsigned)hh < HW) {
        const _Float16* gp = xin + ((size_t)(b * HW + hh) * HW + wstart) * CIN;
        unsigned lds = ldsbase + (unsigned)((r * 66 + (ledge ? 1 : 0)) * CIN * 2);
        // tensor_dim1 clamps at the end of the input row -> right halo beyond
        // w=255 is hardware zero-filled.
        tdm_load_2d(lds, gp, CIN, (unsigned)(HW - wstart), CIN, wcnt,
                    CIN, (unsigned long long)HW * CIN);
      }
    }
    tdm_wait();
  }
#else
  const int NVEC = 3 * 66 * CV;
  for (int v = tid; v < NVEC; v += 256) {
    int r  = v / (66 * CV);
    int rr = v - r * (66 * CV);
    int wi = rr / CV, cv = rr - wi * CV;
    int hh = h + r - 1, ww = w0 - 1 + wi;
    v8h val = {};
    if ((unsigned)hh < HW && (unsigned)ww < HW)
      val = *(const v8h*)(xin + (((size_t)(b * HW + hh) * HW + ww) * CIN + cv * 8));
    *(v8h*)(&sA[v * 8]) = val;
  }
#endif
  __syncthreads();

  const int lane = tid & 31;
  const int wv = tid >> 5;
  const int mwave = wv & 3, nwave = wv >> 2;
  const int mrow = lane & 15, kh = lane >> 4;
  const int pix = mwave * 16 + mrow;   // A-matrix row (pixel) for this lane

  v8f acc[4] = {};
  const int KSTEPS = CIN * 9 / 32;
  const v16h* wfv = (const v16h*)wfrag;

  for (int ks = 0; ks < KSTEPS; ++ks) {
    int k0 = ks * 32;
    int rs = k0 / CIN;             // CIN % 32 == 0 -> one (r,s) tap per k-step
    int r = rs / 3, s = rs - r * 3;
    int cb = k0 - rs * CIN;
    // A fragment: ISA 16-bit A layout -> two contiguous 8-half groups per lane
    const _Float16* ap = &sA[(size_t)(r * 66 + pix + s) * CIN + cb + kh * 8];
    v16h a;
    #pragma unroll
    for (int e = 0; e < 8; ++e) { a[e] = ap[e]; a[e + 8] = ap[e + 16]; }
    if (ks + 2 < KSTEPS)  // keep the weight-fragment stream warm in WGP$
      __builtin_prefetch((const void*)&wfv[(size_t)((ks + 2) * 8 + nwave * 4) * 32 + lane], 0, 1);
    #pragma unroll
    for (int nt = 0; nt < 4; ++nt) {
      int ntg = nwave * 4 + nt;
      v16h bfr = wfv[(size_t)(ks * 8 + ntg) * 32 + lane];
      acc[nt] = __builtin_amdgcn_wmma_f32_16x16x32_f16(
          false, a, false, bfr, (short)0, acc[nt], false, false);
    }
  }

  // Epilogue: per-lane channel n is fixed (f32 D layout), fold BN + ReLU.
  // Stage the tile in LDS and burst it out with one TDM store (16 KB contig).
  #pragma unroll
  for (int nt = 0; nt < 4; ++nt) {
    int n = (nwave * 4 + nt) * 16 + mrow;
    float inv  = gg[n] * rsqrtf(vvar[n] + 1e-5f);
    float bias = bbeta[n] - mmean[n] * inv;
    #pragma unroll
    for (int j = 0; j < 8; ++j) {
      float val = acc[nt][j] * inv + bias;
      if (do_relu) val = fmaxf(val, 0.0f);
#if USE_TDM
      sOut[(mwave * 16 + kh * 8 + j) * C2 + n] = (_Float16)val;
#else
      outp[((size_t)(b * HW + h) * HW + w0 + mwave * 16 + kh * 8 + j) * C2 + n] =
          (_Float16)val;
#endif
    }
  }
#if USE_TDM
  __syncthreads();                                  // LDS tile complete
  if (tid < 32) {
    tdm_store_1d((unsigned)(size_t)&sOut[0],
                 outp + ((size_t)(b * HW + h) * HW + w0) * C2, 64 * C2);
    tdm_wait();                                     // keep LDS alive until done
  }
#endif
}

// ---------------------------------------------------------------------------
// Hermite function stack, r = 6 (hardcoded; harness passes r == 6).
// ---------------------------------------------------------------------------
__device__ inline void hermite6(float x, float* hh) {
  hh[0] = 0.7511255444649425f * __expf(-0.5f * x * x);
  hh[1] = 1.4142135623730951f * x * hh[0];
  hh[2] = 1.0f                * x * hh[1] - 0.7071067811865476f * hh[0];
  hh[3] = 0.816496580927726f  * x * hh[2] - 0.816496580927726f  * hh[1];
  hh[4] = 0.7071067811865476f * x * hh[3] - 0.8660254037844386f * hh[2];
  hh[5] = 0.6324555320336759f * x * hh[4] - 0.8944271909999159f * hh[3];
}

// ---------------------------------------------------------------------------
// Fused tail: sc = bns(conv1x1(x)) via WMMA (K=64), off = relu(h2 + sc) kept
// entirely in LDS (never hits HBM), then Hermite outer product -> d_out.
// x tile (8 KB) and h2 tile (16 KB) are each one contiguous TDM load.
// ---------------------------------------------------------------------------
__global__ __launch_bounds__(256) void shortcut_hermite_kernel(
    const _Float16* __restrict__ xin,    // [B][H][W][64] f16
    const _Float16* __restrict__ wfrag,  // swizzled 1x1 weights (K=64)
    const float* __restrict__ gs, const float* __restrict__ bsb,
    const float* __restrict__ msb, const float* __restrict__ vsb,
    const _Float16* __restrict__ h2,     // [B][H][W][128] f16 (bn2(conv2))
    float* __restrict__ outp)            // [2][64][6][256][256] fp32
{
  __shared__ __align__(16) _Float16 sX[64 * CN];
  __shared__ __align__(16) _Float16 sH2[64 * C2];
  __shared__ __align__(16) float    sOff[64 * C2];

  const int tid = threadIdx.x;
  const int blk = blockIdx.x;
  const int b = blk >> 10, rem = blk & 1023;
  const int h = rem >> 2, w0 = (rem & 3) << 6;

  const _Float16* src   = xin + ((size_t)(b * HW + h) * HW + w0) * CN;
  const _Float16* h2src = h2  + ((size_t)(b * HW + h) * HW + w0) * C2;
#if USE_TDM
  if (tid < 32) {
    tdm_load_2d((unsigned)(size_t)&sX[0],  src,   64 * CN, 1, 64 * CN, 1,
                64 * CN, (unsigned long long)(64 * CN));
    tdm_load_2d((unsigned)(size_t)&sH2[0], h2src, 64 * C2, 1, 64 * C2, 1,
                64 * C2, (unsigned long long)(64 * C2));
    tdm_wait();
  }
#else
  for (int v = tid; v < 64 * CN / 8; v += 256)
    *(v8h*)(&sX[v * 8]) = *(const v8h*)(src + v * 8);
  for (int v = tid; v < 64 * C2 / 8; v += 256)
    *(v8h*)(&sH2[v * 8]) = *(const v8h*)(h2src + v * 8);
#endif
  __syncthreads();

  const int lane = tid & 31;
  const int wv = tid >> 5;
  const int mwave = wv & 3, nwave = wv >> 2;
  const int mrow = lane & 15, kh = lane >> 4;
  const int pix = mwave * 16 + mrow;

  v8f acc[4] = {};
  const v16h* wfv = (const v16h*)wfrag;
  #pragma unroll
  for (int ks = 0; ks < 2; ++ks) {          // K = 64 -> 2 steps of 32
    const _Float16* ap = &sX[pix * 64 + ks * 32 + kh * 8];
    v16h a;
    #pragma unroll
    for (int e = 0; e < 8; ++e) { a[e] = ap[e]; a[e + 8] = ap[e + 16]; }
    #pragma unroll
    for (int nt = 0; nt < 4; ++nt) {
      int ntg = nwave * 4 + nt;
      v16h bfr = wfv[(size_t)(ks * 8 + ntg) * 32 + lane];
      acc[nt] = __builtin_amdgcn_wmma_f32_16x16x32_f16(
          false, a, false, bfr, (short)0, acc[nt], false, false);
    }
  }

  // off = relu(h2 + bn_s(sc)) -> LDS
  #pragma unroll
  for (int nt = 0; nt < 4; ++nt) {
    int n = (nwave * 4 + nt) * 16 + mrow;
    float inv  = gs[n] * rsqrtf(vsb[n] + 1e-5f);
    float bias = bsb[n] - msb[n] * inv;
    #pragma unroll
    for (int j = 0; j < 8; ++j) {
      int prow = mwave * 16 + kh * 8 + j;
      float sc = acc[nt][j] * inv + bias;
      float hv = (float)sH2[prow * C2 + n];
      sOff[prow * C2 + n] = fmaxf(hv + sc, 0.0f);
    }
  }
  __syncthreads();

  // Hermite outer product. thread -> channel c = tid/4, pixel quarter tid%4.
  const int c = tid >> 2, ps = tid & 3;
  const float liny = (float)h * (2.0f / 255.0f) - 1.0f;
  #pragma unroll 4
  for (int i = 0; i < 16; ++i) {
    int p = ps * 16 + i;
    float xg = sOff[p * C2 + c]      + (float)(w0 + p) * (2.0f / 255.0f) - 1.0f;
    float yg = sOff[p * C2 + 64 + c] + liny;
    float hx[6], hy[6];
    hermite6(xg, hx);
    hermite6(yg, hy);
    size_t base = ((size_t)(b * 64 + c) * 6) * 65536 + (size_t)h * 256 + (w0 + p);
    #pragma unroll
    for (int k = 0; k < 6; ++k)
      outp[base + (size_t)k * 65536] = hx[k] * hy[k];
  }
}

// ---------------------------------------------------------------------------
// Launcher.  Workspace layout (~84.3 MB):
//   xh (x as f16 NHWC) | h1 f16 | h2 f16 | w1 frags | w2 frags | ws frags
// ---------------------------------------------------------------------------
extern "C" void kernel_launch(void* const* d_in, const int* in_sizes, int n_in,
                              void* d_out, int out_size, void* d_ws, size_t ws_size,
                              hipStream_t stream) {
  (void)in_sizes; (void)n_in; (void)out_size; (void)ws_size;
  const float* x   = (const float*)d_in[0];
  // d_in[1] == r (always 6; hardcoded to match out_size)
  const float* w1  = (const float*)d_in[2];
  const float* g1  = (const float*)d_in[3];
  const float* b1  = (const float*)d_in[4];
  const float* m1  = (const float*)d_in[5];
  const float* v1  = (const float*)d_in[6];
  const float* w2  = (const float*)d_in[7];
  const float* g2  = (const float*)d_in[8];
  const float* b2  = (const float*)d_in[9];
  const float* m2  = (const float*)d_in[10];
  const float* v2  = (const float*)d_in[11];
  const float* wsw = (const float*)d_in[12];
  const float* gs  = (const float*)d_in[13];
  const float* bs  = (const float*)d_in[14];
  const float* ms  = (const float*)d_in[15];
  const float* vs  = (const float*)d_in[16];
  float* out = (float*)d_out;

  char* p = (char*)d_ws;
  _Float16* xh  = (_Float16*)p; p += (size_t)BATCH * HW * HW * CN * 2;  // 16.8 MB
  _Float16* h1  = (_Float16*)p; p += (size_t)BATCH * HW * HW * C2 * 2;  // 33.6 MB
  _Float16* h2  = (_Float16*)p; p += (size_t)BATCH * HW * HW * C2 * 2;  // 33.6 MB
  _Float16* w1f = (_Float16*)p; p += (size_t)18 * 8 * 32 * 16 * 2;      // 144 KB
  _Float16* w2f = (_Float16*)p; p += (size_t)36 * 8 * 32 * 16 * 2;      // 288 KB
  _Float16* wsf = (_Float16*)p; p += (size_t)2  * 8 * 32 * 16 * 2;      // 16 KB

  prep_x_kernel<<<(BATCH * HW * HW * CN + 255) / 256, 256, 0, stream>>>(x, xh);
  prep_w_kernel<<<(18 * 8 * 32 + 255) / 256, 256, 0, stream>>>(w1, w1f, 64, 9);
  prep_w_kernel<<<(36 * 8 * 32 + 255) / 256, 256, 0, stream>>>(w2, w2f, 128, 9);
  prep_w_kernel<<<(2 * 8 * 32 + 255) / 256, 256, 0, stream>>>(wsw, wsf, 64, 1);

  dim3 grid(BATCH * HW * (HW / 64)), blk(256);   // 2048 tiles of 64 px x 128 ch
  conv3x3_bn_kernel<64 ><<<grid, blk, 0, stream>>>(xh, w1f, g1, b1, m1, v1, h1, 1);
  conv3x3_bn_kernel<128><<<grid, blk, 0, stream>>>(h1, w2f, g2, b2, m2, v2, h2, 0);
  shortcut_hermite_kernel<<<grid, blk, 0, stream>>>(xh, wsf, gs, bs, ms, vs, h2, out);
}
